// NNS_65652870087352
// MI455X (gfx1250) — matrix operand order repaired and checked by reading, not compiled, not run on hardware
//
#include <hip/hip_runtime.h>

typedef __attribute__((ext_vector_type(16))) _Float16 v16h;
typedef __attribute__((ext_vector_type(8)))  float    v8f;
typedef unsigned int u32;
typedef __attribute__((ext_vector_type(4))) u32 u32x4;
typedef __attribute__((ext_vector_type(8))) int i32x8;
typedef __attribute__((ext_vector_type(4))) int i32x4;

namespace {
constexpr int Bn = 1024, Tn = 128, Dn = 128, Hn = 8, Ln = 2, FFn = 512;
constexpr int D3 = 3 * Dn;

// LDS layout (bytes)
constexpr int OFF_XF32 = 0;        // [128][128] f32   64 KB  residual stream
constexpr int OFF_XF16 = 65536;    // [128][128] f16   32 KB  f16 x copy / attn-out
constexpr int OFF_QKV  = 98304;    // [128][384] f16   96 KB  q|k|v
constexpr int OFF_SCR  = 196608;   // 96 KB scratch: weights / scores / p / h1 / f32 temps
constexpr int OFF_MASK = 294912;   // [128] f32 additive key mask
constexpr int SMEM_BYTES = 295424; // 288.5 KB < 320 KB WGP limit
}

// ---------------- Tensor Data Mover: 128x128 f32 tile, global -> LDS ----------------
// D# packing per CDNA5 ISA:
//  group0: [1:0]count=1 | [63:32]lds_addr | [120:64]global_addr | [127:126]type=2
//  group1: [17:16]data_size=2(4B) | [79:48]tensor_dim0 | [111:80]tensor_dim1
//          | [127:112]tile_dim0 | [143:128]tile_dim1 | [207:160]tensor_dim0_stride
// Toolchain is the 6-arg clang-23 form: (u32x4, i32x8, i32x4, i32x4, i32x8, i32 cpol)
__device__ __forceinline__ void tdm_load_128x128_f32(const float* g, void* l) {
  unsigned long long ga = (unsigned long long)(uintptr_t)g;
  u32 lds = (u32)(uintptr_t)l;  // flat LDS aperture: low 32 bits = LDS byte offset
  u32x4 g0;
  g0[0] = 1u;                                   // count=1, user descriptor
  g0[1] = lds;                                  // lds_addr
  g0[2] = (u32)ga;                              // global_addr[31:0]
  g0[3] = (u32)((ga >> 32) & 0x01ffffffu) | (2u << 30);  // addr[56:32] | type=2
  i32x8 g1;
  g1[0] = 0x20000;            // data_size = 2 (4 bytes), no multicast/pad/iterate
  g1[1] = (int)(128u << 16);  // tensor_dim0 = 128   (bits 79:48)
  g1[2] = (int)(128u << 16);  // tensor_dim1 = 128   (bits 111:80)
  g1[3] = (int)(128u << 16);  // tile_dim0   = 128   (bits 127:112)
  g1[4] = 128;                // tile_dim1   = 128, tile_dim2 = 0
  g1[5] = 128;                // tensor_dim0_stride = 128
  g1[6] = 0;
  g1[7] = 0;
  i32x4 z4;
  z4[0] = 0; z4[1] = 0; z4[2] = 0; z4[3] = 0;
  i32x8 z8;
#pragma unroll
  for (int i = 0; i < 8; ++i) z8[i] = 0;
  __builtin_amdgcn_tensor_load_to_lds(g0, g1, z4, z4, z8, 0);
}

// ---- WMMA fragment loaders, contiguous k order (k = 16*half + s) in both A and B ----

// A-matrix 16x32 f16: lane = M + 16*half; VGPR p holds K pair
__device__ __forceinline__ v16h ld_a16(const _Float16* s, int stride, int r0, int c0, int lane) {
  int m = lane & 15, hh = lane >> 4;
  v16h a;
#pragma unroll
  for (int p = 0; p < 8; ++p) {
    int kb = (hh << 4) + (p << 1);
    const _Float16* q = s + (r0 + m) * stride + c0 + kb;
    a[2 * p] = q[0];
    a[2 * p + 1] = q[1];
  }
  return a;
}

// A load with only K=0..15 valid (HD=16 zero-padded to 32)
__device__ __forceinline__ v16h ld_a16_pad16(const _Float16* s, int stride, int r0, int c0, int lane) {
  int m = lane & 15, hh = lane >> 4;
  v16h a;
#pragma unroll
  for (int p = 0; p < 8; ++p) {
    int kb = (hh << 4) + (p << 1);
    if (kb < 16) {
      const _Float16* q = s + (r0 + m) * stride + c0 + kb;
      a[2 * p] = q[0];
      a[2 * p + 1] = q[1];
    } else {
      a[2 * p] = (_Float16)0.f;
      a[2 * p + 1] = (_Float16)0.f;
    }
  }
  return a;
}

// B-matrix 32x16 f16 from transposed source (row-major [N][K]): lane = N + 16*Khalf
__device__ __forceinline__ v16h ld_bt16(const _Float16* s, int stride, int n0, int k0, int lane) {
  int n = lane & 15, hh = lane >> 4;
  v16h b;
#pragma unroll
  for (int p = 0; p < 8; ++p) {
    int kb = (hh << 4) + (p << 1);
    const _Float16* q = s + (n0 + n) * stride + k0 + kb;
    b[2 * p] = q[0];
    b[2 * p + 1] = q[1];
  }
  return b;
}

__device__ __forceinline__ v16h ld_bt16_pad16(const _Float16* s, int stride, int n0, int k0, int lane) {
  int n = lane & 15, hh = lane >> 4;
  v16h b;
#pragma unroll
  for (int p = 0; p < 8; ++p) {
    int kb = (hh << 4) + (p << 1);
    if (kb < 16) {
      const _Float16* q = s + (n0 + n) * stride + k0 + kb;
      b[2 * p] = q[0];
      b[2 * p + 1] = q[1];
    } else {
      b[2 * p] = (_Float16)0.f;
      b[2 * p + 1] = (_Float16)0.f;
    }
  }
  return b;
}

// B-matrix 32x16 f16 from natural row-major [K][N] source (V matrix: rows=tokens=K)
__device__ __forceinline__ v16h ld_bn16(const _Float16* s, int stride, int k0, int n0, int lane) {
  int n = lane & 15, hh = lane >> 4;
  v16h b;
#pragma unroll
  for (int p = 0; p < 8; ++p) {
    int kb = (hh << 4) + (p << 1);
    b[2 * p]     = s[(k0 + kb) * stride + n0 + n];
    b[2 * p + 1] = s[(k0 + kb + 1) * stride + n0 + n];
  }
  return b;
}

__device__ __forceinline__ v8f zero8() {
  v8f z;
#pragma unroll
  for (int i = 0; i < 8; ++i) z[i] = 0.f;
  return z;
}

__device__ __forceinline__ v8f wmma_f16(v16h a, v16h b, v8f c) {
  return __builtin_amdgcn_wmma_f32_16x16x32_f16(false, a, false, b, (short)0, c, false, false);
}

// residual add + LayerNorm, 2 threads per row, float4 + wave32 shfl combine
__device__ __forceinline__ void resid_ln(float* xf32, _Float16* xf16, const float* add,
                                         const float* g, const float* be, int tid) {
  const int row = tid >> 1, hp = tid & 1;
  float4* x4 = (float4*)(xf32 + row * Dn + hp * 64);
  const float4* a4 = (const float4*)(add + row * Dn + hp * 64);
  float msum = 0.f;
#pragma unroll
  for (int j = 0; j < 16; ++j) {
    float4 xv = x4[j], av = a4[j];
    xv.x += av.x; xv.y += av.y; xv.z += av.z; xv.w += av.w;
    x4[j] = xv;
    msum += xv.x + xv.y + xv.z + xv.w;
  }
  msum += __shfl_xor(msum, 1, 32);
  float mean = msum * (1.f / Dn);
  float vsum = 0.f;
#pragma unroll
  for (int j = 0; j < 16; ++j) {
    float4 s = x4[j];
    float d0 = s.x - mean, d1 = s.y - mean, d2 = s.z - mean, d3 = s.w - mean;
    vsum += d0 * d0 + d1 * d1 + d2 * d2 + d3 * d3;
  }
  vsum += __shfl_xor(vsum, 1, 32);
  float inv = rsqrtf(vsum * (1.f / Dn) + 1e-5f);
  _Float16* xh = xf16 + row * Dn + hp * 64;
#pragma unroll
  for (int j = 0; j < 16; ++j) {
    float4 s = x4[j];
    int col = hp * 64 + 4 * j;
    float v0 = (s.x - mean) * inv * g[col + 0] + be[col + 0];
    float v1 = (s.y - mean) * inv * g[col + 1] + be[col + 1];
    float v2 = (s.z - mean) * inv * g[col + 2] + be[col + 2];
    float v3 = (s.w - mean) * inv * g[col + 3] + be[col + 3];
    x4[j] = make_float4(v0, v1, v2, v3);
    xh[4 * j + 0] = (_Float16)v0; xh[4 * j + 1] = (_Float16)v1;
    xh[4 * j + 2] = (_Float16)v2; xh[4 * j + 3] = (_Float16)v3;
  }
}

__global__ __launch_bounds__(256, 1)
void NNS_fused_transformer(const float* __restrict__ xg,
                           const float* __restrict__ attn_mask,
                           const float* __restrict__ in_proj_w,
                           const float* __restrict__ in_proj_b,
                           const float* __restrict__ out_w,
                           const float* __restrict__ out_b,
                           const float* __restrict__ ln1_g,
                           const float* __restrict__ ln1_b,
                           const float* __restrict__ ff1_w,
                           const float* __restrict__ ff1_b,
                           const float* __restrict__ ff2_w,
                           const float* __restrict__ ff2_b,
                           const float* __restrict__ ln2_g,
                           const float* __restrict__ ln2_b,
                           float* __restrict__ outg) {
  extern __shared__ unsigned char smem[];
  float*    xf32    = (float*)(smem + OFF_XF32);
  _Float16* xf16    = (_Float16*)(smem + OFF_XF16);  // reused as per-head attn output
  _Float16* qkv16   = (_Float16*)(smem + OFF_QKV);
  float*    scr_f32 = (float*)(smem + OFF_SCR);
  _Float16* scr_f16 = (_Float16*)(smem + OFF_SCR);
  float*    stage32 = (float*)(smem + OFF_SCR + 32768); // 64 KB f32 TDM landing zone
  float*    maskb   = (float*)(smem + OFF_MASK);

  const int b    = blockIdx.x;
  const int tid  = threadIdx.x;
  const int wave = tid >> 5;
  const int lane = tid & 31;
  const int m0   = wave * 16;            // this wave's M-slab
  const int cn   = lane & 15;            // C-fragment column within tile
  const int mb   = (lane >> 4) << 3;     // C-fragment row base (M = mb + r)

  // ---- x[b] via Tensor Data Mover, then f16 copy ----
  if (wave == 0) {
    tdm_load_128x128_f32(xg + (size_t)b * Tn * Dn, xf32);
    __builtin_amdgcn_s_wait_tensorcnt(0);
  }
  for (int i = tid; i < Tn; i += 256) maskb[i] = attn_mask[(size_t)b * Tn + i];
  __syncthreads();
  for (int i = tid; i < Tn * Dn / 4; i += 256) {
    float4 v = ((const float4*)xf32)[i];
    _Float16* d16 = xf16 + 4 * i;
    d16[0] = (_Float16)v.x; d16[1] = (_Float16)v.y;
    d16[2] = (_Float16)v.z; d16[3] = (_Float16)v.w;
  }
  __syncthreads();

  for (int l = 0; l < Ln; ++l) {
    const float* Wqkv = in_proj_w + (size_t)l * D3 * Dn;
    const float* bq   = in_proj_b + (size_t)l * D3;
    const float* Wo   = out_w + (size_t)l * Dn * Dn;
    const float* bo   = out_b + (size_t)l * Dn;
    const float* g1   = ln1_g + (size_t)l * Dn;
    const float* be1  = ln1_b + (size_t)l * Dn;
    const float* W1   = ff1_w + (size_t)l * FFn * Dn;
    const float* b1   = ff1_b + (size_t)l * FFn;
    const float* W2   = ff2_w + (size_t)l * Dn * FFn;
    const float* b2   = ff2_b + (size_t)l * Dn;
    const float* g2   = ln2_g + (size_t)l * Dn;
    const float* be2  = ln2_b + (size_t)l * Dn;

    // ================= QKV projection: qkv = x @ Wqkv^T + bq =================
    {
      const float4* wg = (const float4*)Wqkv;      // stage 384x128 weights as f16
      for (int i = tid; i < D3 * Dn / 4; i += 256) {
        float4 v = wg[i];
        _Float16* d = scr_f16 + 4 * i;
        d[0] = (_Float16)v.x; d[1] = (_Float16)v.y;
        d[2] = (_Float16)v.z; d[3] = (_Float16)v.w;
      }
      if (l + 1 < Ln) // warm next layer's weights into L2 (global_prefetch_b8)
        __builtin_prefetch(Wqkv + (size_t)D3 * Dn + tid * 32, 0, 1);
    }
    __syncthreads();
    {
      v16h ax[4];
#pragma unroll
      for (int ks = 0; ks < 4; ++ks) ax[ks] = ld_a16(xf16, Dn, m0, ks * 32, lane);
      for (int nt = 0; nt < D3 / 16; ++nt) {
        v8f acc = zero8();
#pragma unroll
        for (int ks = 0; ks < 4; ++ks)
          acc = wmma_f16(ax[ks], ld_bt16(scr_f16, Dn, nt * 16, ks * 32, lane), acc);
        int col = nt * 16 + cn;
        float bb = bq[col];
#pragma unroll
        for (int r = 0; r < 8; ++r)
          qkv16[(m0 + mb + r) * D3 + col] = (_Float16)(acc[r] + bb);
      }
    }
    __syncthreads();

    // ================= attention (head-serial, full-block GEMMs) =================
    float*    scores = scr_f32;                             // [128][128] f32
    _Float16* p16    = (_Float16*)(smem + OFF_SCR + 65536); // [128][128] f16
    for (int h = 0; h < Hn; ++h) {
      // S = Q_h K_h^T * scale + mask  (HD=16 zero-padded to K=32)
      v16h aq = ld_a16_pad16(qkv16, D3, m0, h * 16, lane);
      for (int nt = 0; nt < 8; ++nt) {
        v16h bk = ld_bt16_pad16(qkv16, D3, nt * 16, Dn + h * 16, lane);
        v8f acc = wmma_f16(aq, bk, zero8());
        int col = nt * 16 + cn;
        float mk = maskb[col];
#pragma unroll
        for (int r = 0; r < 8; ++r)
          scores[(m0 + mb + r) * Tn + col] = acc[r] * 0.25f + mk;
      }
      __syncthreads();
      // row softmax -> p16 : 2 threads per row, float4, shfl combine
      {
        const int row = tid >> 1, hp = tid & 1;
        float4* s4 = (float4*)(scores + row * Tn + hp * 64);
        float mx = -3.4e38f;
#pragma unroll
        for (int j = 0; j < 16; ++j) {
          float4 v = s4[j];
          mx = fmaxf(mx, fmaxf(fmaxf(v.x, v.y), fmaxf(v.z, v.w)));
        }
        mx = fmaxf(mx, __shfl_xor(mx, 1, 32));
        float sum = 0.f;
#pragma unroll
        for (int j = 0; j < 16; ++j) {
          float4 v = s4[j];
          v.x = __expf(v.x - mx); v.y = __expf(v.y - mx);
          v.z = __expf(v.z - mx); v.w = __expf(v.w - mx);
          s4[j] = v;
          sum += v.x + v.y + v.z + v.w;
        }
        sum += __shfl_xor(sum, 1, 32);
        float inv = 1.f / sum;
        _Float16* pr = p16 + row * Tn + hp * 64;
#pragma unroll
        for (int j = 0; j < 16; ++j) {
          float4 v = s4[j];
          pr[4 * j + 0] = (_Float16)(v.x * inv);
          pr[4 * j + 1] = (_Float16)(v.y * inv);
          pr[4 * j + 2] = (_Float16)(v.z * inv);
          pr[4 * j + 3] = (_Float16)(v.w * inv);
        }
      }
      __syncthreads();
      // O_h = P @ V_h -> columns [16h,16h+16) of attn-out (over xf16, dead this phase)
      {
        v8f acc = zero8();
#pragma unroll
        for (int ks = 0; ks < 4; ++ks) {
          v16h ap = ld_a16(p16, Tn, m0, ks * 32, lane);
          v16h bv = ld_bn16(qkv16, D3, ks * 32, 2 * Dn + h * 16, lane);
          acc = wmma_f16(ap, bv, acc);
        }
#pragma unroll
        for (int r = 0; r < 8; ++r)
          xf16[(m0 + mb + r) * Dn + h * 16 + cn] = (_Float16)acc[r];
      }
      __syncthreads();
    }

    // ================= output projection + residual + LN1 =================
    if (wave == 0) { // TDM: out_w f32 -> landing zone
      tdm_load_128x128_f32(Wo, stage32);
      __builtin_amdgcn_s_wait_tensorcnt(0);
    }
    __syncthreads();
    for (int i = tid; i < Dn * Dn / 4; i += 256) { // convert to f16 at scratch[0,32K)
      float4 v = ((const float4*)stage32)[i];
      _Float16* d = scr_f16 + 4 * i;
      d[0] = (_Float16)v.x; d[1] = (_Float16)v.y;
      d[2] = (_Float16)v.z; d[3] = (_Float16)v.w;
    }
    __syncthreads();
    float* attnout = stage32; // [128][128] f32 (reuse landing zone)
    {
      v16h ao[4];
#pragma unroll
      for (int ks = 0; ks < 4; ++ks) ao[ks] = ld_a16(xf16, Dn, m0, ks * 32, lane);
      for (int nt = 0; nt < 8; ++nt) {
        v8f acc = zero8();
#pragma unroll
        for (int ks = 0; ks < 4; ++ks)
          acc = wmma_f16(ao[ks], ld_bt16(scr_f16, Dn, nt * 16, ks * 32, lane), acc);
        int col = nt * 16 + cn;
        float bb = bo[col];
#pragma unroll
        for (int r = 0; r < 8; ++r)
          attnout[(m0 + mb + r) * Dn + col] = acc[r] + bb;
      }
    }
    __syncthreads();
    resid_ln(xf32, xf16, attnout, g1, be1, tid);
    __syncthreads();

    // ================= feed-forward, K-chunked (4 x 128), accum in VGPRs =================
    _Float16* w1buf = scr_f16;                              // [128][128] f16 [0,32K)
    _Float16* w2buf = (_Float16*)(smem + OFF_SCR + 32768);  // [128][128] f16 [32K,64K)
    _Float16* h1buf = (_Float16*)(smem + OFF_SCR + 65536);  // [128][128] f16 [64K,96K)
    v8f acc2[8];
#pragma unroll
    for (int nt = 0; nt < 8; ++nt) acc2[nt] = zero8();
    v16h axf[4];
#pragma unroll
    for (int ks = 0; ks < 4; ++ks) axf[ks] = ld_a16(xf16, Dn, m0, ks * 32, lane);

    for (int c = 0; c < 4; ++c) {
      if (wave == 0) { // TDM: ff1 chunk (contiguous 128x128 f32) -> landing zone
        tdm_load_128x128_f32(W1 + (size_t)c * Dn * Dn, stage32);
        __builtin_amdgcn_s_wait_tensorcnt(0);
      }
      __syncthreads();
      for (int i = tid; i < Dn * Dn / 4; i += 256) { // convert w1 -> f16 [0,32K)
        float4 v = ((const float4*)stage32)[i];
        _Float16* d = w1buf + 4 * i;
        d[0] = (_Float16)v.x; d[1] = (_Float16)v.y;
        d[2] = (_Float16)v.z; d[3] = (_Float16)v.w;
      }
      __syncthreads();
      for (int i = tid; i < Dn * Dn; i += 256) { // stage ff2 chunk cols (strided) -> f16
        int r = i >> 7, cc = i & 127;
        w2buf[i] = (_Float16)W2[(size_t)r * FFn + c * Dn + cc];
      }
      __syncthreads();
      // h1_c = relu(x @ W1c^T + b1c)
      for (int nt = 0; nt < 8; ++nt) {
        v8f acc = zero8();
#pragma unroll
        for (int ks = 0; ks < 4; ++ks)
          acc = wmma_f16(axf[ks], ld_bt16(w1buf, Dn, nt * 16, ks * 32, lane), acc);
        int col = nt * 16 + cn;
        float bb = b1[c * Dn + col];
#pragma unroll
        for (int r = 0; r < 8; ++r)
          h1buf[(m0 + mb + r) * Dn + col] = (_Float16)fmaxf(acc[r] + bb, 0.f);
      }
      __syncthreads();
      // acc2 += h1_c @ W2c^T
      {
        v16h ah[4];
#pragma unroll
        for (int ks = 0; ks < 4; ++ks) ah[ks] = ld_a16(h1buf, Dn, m0, ks * 32, lane);
        for (int nt = 0; nt < 8; ++nt) {
#pragma unroll
          for (int ks = 0; ks < 4; ++ks)
            acc2[nt] = wmma_f16(ah[ks], ld_bt16(w2buf, Dn, nt * 16, ks * 32, lane), acc2[nt]);
        }
      }
      __syncthreads();
    }
    // spill FF output, then residual + LN2
    float* ffout = scr_f32;
    for (int nt = 0; nt < 8; ++nt) {
      int col = nt * 16 + cn;
      float bb = b2[col];
#pragma unroll
      for (int r = 0; r < 8; ++r)
        ffout[(m0 + mb + r) * Dn + col] = acc2[nt][r] + bb;
    }
    __syncthreads();
    resid_ln(xf32, xf16, ffout, g2, be2, tid);
    __syncthreads();
  }

  // ---- write result ----
  {
    float4* dst = (float4*)(outg + (size_t)b * Tn * Dn);
    const float4* src = (const float4*)xf32;
    for (int i = tid; i < Tn * Dn / 4; i += 256) dst[i] = src[i];
  }
}

extern "C" void kernel_launch(void* const* d_in, const int* in_sizes, int n_in,
                              void* d_out, int out_size, void* d_ws, size_t ws_size,
                              hipStream_t stream) {
  (void)in_sizes; (void)n_in; (void)out_size; (void)d_ws; (void)ws_size;
  const float* x     = (const float*)d_in[0];
  const float* mask  = (const float*)d_in[1];
  const float* ipw   = (const float*)d_in[2];
  const float* ipb   = (const float*)d_in[3];
  const float* ow    = (const float*)d_in[4];
  const float* ob    = (const float*)d_in[5];
  const float* g1    = (const float*)d_in[6];
  const float* be1   = (const float*)d_in[7];
  const float* f1w   = (const float*)d_in[8];
  const float* f1b   = (const float*)d_in[9];
  const float* f2w   = (const float*)d_in[10];
  const float* f2b   = (const float*)d_in[11];
  const float* g2    = (const float*)d_in[12];
  const float* be2   = (const float*)d_in[13];
  float* out = (float*)d_out;

  hipError_t err = hipFuncSetAttribute((const void*)NNS_fused_transformer,
                                       hipFuncAttributeMaxDynamicSharedMemorySize, SMEM_BYTES);
  (void)err;
  NNS_fused_transformer<<<Bn, 256, SMEM_BYTES, stream>>>(
      x, mask, ipw, ipb, ow, ob, g1, be1, f1w, f1b, f2w, f2b, g2, be2, out);
}
